// Llama4Attention_20684562497548
// MI455X (gfx1250) — matrix operand order repaired and checked by reading, not compiled, not run on hardware
//
#include <hip/hip_runtime.h>
#include <stdint.h>

// ---------------- problem constants ----------------
#define S_LEN 2048
#define HDIM  5120
#define NH    40
#define NKV   8
#define HD    128
#define QKVW  ((NH + 2 * NKV) * HD)   // 7168
#define GQA   (NH / NKV)              // 5
#define THETA 500000.0f
#define EPS_R 1e-5f
#define SCALE 0.08838834764831845f    // 128^-0.5

// ---------------- types ----------------
typedef __bf16 bf16_t;
typedef __attribute__((ext_vector_type(16))) __bf16 v16bf;
typedef __attribute__((ext_vector_type(8)))  __bf16 v8bf;
typedef __attribute__((ext_vector_type(4)))  __bf16 v4bf;
typedef __attribute__((ext_vector_type(8)))  float  v8f;
typedef __attribute__((ext_vector_type(4)))  float  v4f;

__device__ __forceinline__ bf16_t to_bf16(float f) {
    uint32_t u = __builtin_bit_cast(uint32_t, f);
    uint32_t r = u + 0x7FFFu + ((u >> 16) & 1u);   // RNE
    return __builtin_bit_cast(bf16_t, (uint16_t)(r >> 16));
}

__device__ __forceinline__ v16bf mk16(v8bf a, v8bf b) {
    v16bf r;
#pragma unroll
    for (int i = 0; i < 8; ++i) { r[i] = a[i]; r[i + 8] = b[i]; }
    return r;
}

__device__ __forceinline__ v8f wmma_bf16(v16bf a, v16bf b, v8f c) {
    return __builtin_amdgcn_wmma_f32_16x16x32_bf16(
        false, a, false, b, (short)0, c, false, false);
}

// ---- gfx1250 async global->LDS (ASYNCcnt-tracked) ----
__device__ __forceinline__ void async_load_b128(uint32_t lds_off, const void* gaddr) {
    asm volatile("global_load_async_to_lds_b128 %0, %1, off"
                 :: "v"(lds_off), "v"((uint64_t)(uintptr_t)gaddr) : "memory");
}
__device__ __forceinline__ void wait_async0() {
    asm volatile("s_wait_asynccnt 0x0" ::: "memory");
}

// =====================================================================
// GEMM: C[M][N] = A[M][K] * B[N][K]^T ; A f32 or bf16, B f32, C f32.
// 128x128 block, BK=32, double-buffered LDS, 8 waves (2M x 4N),
// each wave 4x2 tiles of 16x16 WMMA. bf16-A path stages A via
// global_load_async_to_lds_b128.
// =====================================================================
template <bool A_F32>
__global__ __launch_bounds__(256)
void gemm_kernel(const void* __restrict__ Ap, const float* __restrict__ B,
                 float* __restrict__ C, int K, int N) {
    __shared__ __align__(16) bf16_t As[2][128 * 40];  // padded stride 40 (80B)
    __shared__ __align__(16) bf16_t Bs[2][128 * 40];

    const int bm = blockIdx.y, bn = blockIdx.x;
    const int t = threadIdx.x;
    const int lane = t & 31;
    const int w = t >> 5;
    const int wm = w >> 2, wn = w & 3;
    const int half = lane >> 4, ln = lane & 15;

    // f32 staging geometry: 4 x (row, 4 floats)
    const int row4 = t >> 3;             // 0..31 (+ i*32)
    const int kk4  = (t & 7) << 2;       // 0..28
    // async bf16-A staging geometry: 2 x (row, 8 bf16)
    const int rowA = t >> 2;             // 0..63 (+ i*64)
    const int kkA  = (t & 3) << 3;       // 0..24

    v8f acc[4][2];
#pragma unroll
    for (int a = 0; a < 4; ++a)
#pragma unroll
        for (int b = 0; b < 2; ++b)
#pragma unroll
            for (int r = 0; r < 8; ++r) acc[a][b][r] = 0.0f;

    const int nk = K >> 5;
    v4f areg[4], breg[4];

    auto load_regs = [&](int it) {
        const int k0 = it << 5;
#pragma unroll
        for (int i = 0; i < 4; ++i) {
            const int row = row4 + i * 32;
            if constexpr (A_F32)
                areg[i] = *(const v4f*)((const float*)Ap +
                                        (size_t)(bm * 128 + row) * K + k0 + kk4);
            const float* bptr = B + (size_t)(bn * 128 + row) * K + k0 + kk4;
            breg[i] = *(const v4f*)bptr;
            if (k0 + 64 < K) __builtin_prefetch((const void*)(bptr + 32), 0, 1);
        }
    };
    auto issue_async_A = [&](int it, int buf) {
        const int k0 = it << 5;
        const bf16_t* Ab = (const bf16_t*)Ap;
#pragma unroll
        for (int i = 0; i < 2; ++i) {
            const int row = rowA + i * 64;
            uint32_t loff = (uint32_t)(uintptr_t)&As[buf][row * 40 + kkA];
            async_load_b128(loff, Ab + (size_t)(bm * 128 + row) * K + k0 + kkA);
        }
    };

    load_regs(0);
    if constexpr (!A_F32) issue_async_A(0, 0);

    for (int it = 0; it < nk; ++it) {
        const int cur = it & 1;
        // ---- store staged regs into LDS (convert f32 -> bf16) ----
#pragma unroll
        for (int i = 0; i < 4; ++i) {
            const int row = row4 + i * 32;
            if constexpr (A_F32) {
                v4bf pa;
                pa[0] = to_bf16(areg[i][0]); pa[1] = to_bf16(areg[i][1]);
                pa[2] = to_bf16(areg[i][2]); pa[3] = to_bf16(areg[i][3]);
                *(v4bf*)&As[cur][row * 40 + kk4] = pa;
            }
            v4bf pb;
            pb[0] = to_bf16(breg[i][0]); pb[1] = to_bf16(breg[i][1]);
            pb[2] = to_bf16(breg[i][2]); pb[3] = to_bf16(breg[i][3]);
            *(v4bf*)&Bs[cur][row * 40 + kk4] = pb;
        }
        const bool has_next = (it + 1 < nk);
        if (has_next) load_regs(it + 1);
        if constexpr (!A_F32) wait_async0();   // tile-it async A is in LDS
        __syncthreads();
        // safe now: everyone finished reading buffer cur^1 (iteration it-1)
        if constexpr (!A_F32) {
            if (has_next) issue_async_A(it + 1, cur ^ 1);
        }

        // ---- compute from LDS[cur] ----
        v16bf bfr[2];
#pragma unroll
        for (int nt = 0; nt < 2; ++nt) {
            const int n = wn * 32 + nt * 16 + ln;
            v8bf b0 = *(const v8bf*)&Bs[cur][n * 40 + half * 8];
            v8bf b1 = *(const v8bf*)&Bs[cur][n * 40 + 16 + half * 8];
            bfr[nt] = mk16(b0, b1);
        }
#pragma unroll
        for (int mt = 0; mt < 4; ++mt) {
            const int m = wm * 64 + mt * 16 + ln;
            v8bf a0 = *(const v8bf*)&As[cur][m * 40 + half * 8];
            v8bf a1 = *(const v8bf*)&As[cur][m * 40 + 16 + half * 8];
            v16bf af = mk16(a0, a1);
            acc[mt][0] = wmma_bf16(af, bfr[0], acc[mt][0]);
            acc[mt][1] = wmma_bf16(af, bfr[1], acc[mt][1]);
        }
    }

    // ---- epilogue: C layout M = r + 8*half, N = ln ----
#pragma unroll
    for (int mt = 0; mt < 4; ++mt)
#pragma unroll
        for (int nt = 0; nt < 2; ++nt) {
            const int col = bn * 128 + wn * 32 + nt * 16 + ln;
#pragma unroll
            for (int r = 0; r < 8; ++r) {
                const int rowg = bm * 128 + wm * 64 + mt * 16 + r + half * 8;
                C[(size_t)rowg * N + col] = acc[mt][nt][r];
            }
        }
}

// =====================================================================
// RoPE (neox) + per-head QK-RMSNorm (q pre-scaled by D^-0.5).
//   q -> qb[NH][S][HD], k -> kb[NKV][S][HD], v -> vt[NKV][HD][S]
// =====================================================================
__global__ __launch_bounds__(256)
void rope_norm_kernel(const int* __restrict__ positions,
                      const float* __restrict__ qkv,
                      const float* __restrict__ w_qk,
                      bf16_t* __restrict__ qb, bf16_t* __restrict__ kb,
                      bf16_t* __restrict__ vt) {
    const int s = blockIdx.x;
    const int w = threadIdx.x >> 5, lane = threadIdx.x & 31;
    const float pos = (float)positions[s];
    const float* row = qkv + (size_t)s * QKVW;

    float inv0 = __powf(THETA, -(2.0f * (float)lane) / 128.0f);
    float inv1 = __powf(THETA, -(2.0f * (float)(lane + 32)) / 128.0f);
    float s0, c0, s1, c1;
    __sincosf(pos * inv0, &s0, &c0);
    __sincosf(pos * inv1, &s1, &c1);
    const float w0 = w_qk[lane],      w1 = w_qk[lane + 32];
    const float w2 = w_qk[lane + 64], w3 = w_qk[lane + 96];

#pragma unroll
    for (int c = 0; c < 7; ++c) {
        const int hh = w * 7 + c;
        const float* hp = row + hh * HD;
        float x0 = hp[lane],      x1 = hp[lane + 32];
        float x2 = hp[lane + 64], x3 = hp[lane + 96];
        if (hh < NH + NKV) {
            float r0 = x0 * c0 - x2 * s0;
            float r2 = x2 * c0 + x0 * s0;
            float r1 = x1 * c1 - x3 * s1;
            float r3 = x3 * c1 + x1 * s1;
            float ss = r0 * r0 + r1 * r1 + r2 * r2 + r3 * r3;
#pragma unroll
            for (int m = 16; m >= 1; m >>= 1) ss += __shfl_xor(ss, m, 32);
            float inv = __frsqrt_rn(ss * (1.0f / 128.0f) + EPS_R);
            if (hh < NH) inv *= SCALE;            // fold D^-0.5 into q
            r0 *= inv * w0; r1 *= inv * w1; r2 *= inv * w2; r3 *= inv * w3;
            bf16_t* dst = (hh < NH)
                ? qb + ((size_t)hh * S_LEN + s) * HD
                : kb + ((size_t)(hh - NH) * S_LEN + s) * HD;
            dst[lane]      = to_bf16(r0);
            dst[lane + 32] = to_bf16(r1);
            dst[lane + 64] = to_bf16(r2);
            dst[lane + 96] = to_bf16(r3);
        } else {
            bf16_t* dst = vt + (size_t)(hh - NH - NKV) * HD * S_LEN;
            dst[(size_t)(lane)      * S_LEN + s] = to_bf16(x0);
            dst[(size_t)(lane + 32) * S_LEN + s] = to_bf16(x1);
            dst[(size_t)(lane + 64) * S_LEN + s] = to_bf16(x2);
            dst[(size_t)(lane + 96) * S_LEN + s] = to_bf16(x3);
        }
    }
}

// =====================================================================
// Flash attention tile: 16 q rows x 32 kv cols; MASKED only on the
// single diagonal tile per wave (no compares in the steady-state loop).
// =====================================================================
template <bool MASKED>
__device__ __forceinline__ void attn_tile(
    int j0, int i0, int half, int ln,
    const bf16_t* __restrict__ kbh, const bf16_t* __restrict__ vth,
    bf16_t* __restrict__ Pw, const v16bf (&qf)[4],
    v8f (&o)[8], float (&mstat)[8], float (&lstat)[8]) {
    // ---- S = Q K^T ----
    v8f sacc[2];
#pragma unroll
    for (int jt = 0; jt < 2; ++jt) {
#pragma unroll
        for (int r = 0; r < 8; ++r) sacc[jt][r] = 0.0f;
        const bf16_t* krow = kbh + (size_t)(j0 + jt * 16 + ln) * HD;
#pragma unroll
        for (int kc = 0; kc < 4; ++kc) {
            v8bf b0 = *(const v8bf*)(krow + kc * 32 + half * 8);
            v8bf b1 = *(const v8bf*)(krow + kc * 32 + 16 + half * 8);
            sacc[jt] = wmma_bf16(qf[kc], mk16(b0, b1), sacc[jt]);
        }
    }
    // ---- online softmax update ----
#pragma unroll
    for (int r = 0; r < 8; ++r) {
        float sv0 = sacc[0][r];
        float sv1 = sacc[1][r];
        if (MASKED) {
            const int i = i0 + r + half * 8;
            if (j0 + ln > i)      sv0 = -1e30f;
            if (j0 + 16 + ln > i) sv1 = -1e30f;
        }
        float mx = fmaxf(sv0, sv1);
#pragma unroll
        for (int mm = 8; mm >= 1; mm >>= 1)
            mx = fmaxf(mx, __shfl_xor(mx, mm, 32));
        float newm = fmaxf(mstat[r], mx);
        float p0 = __expf(sv0 - newm), p1 = __expf(sv1 - newm);
        float rs = p0 + p1;
#pragma unroll
        for (int mm = 8; mm >= 1; mm >>= 1) rs += __shfl_xor(rs, mm, 32);
        float corr = __expf(mstat[r] - newm);
        mstat[r] = newm;
        lstat[r] = lstat[r] * corr + rs;
#pragma unroll
        for (int t = 0; t < 8; ++t) o[t][r] *= corr;
        const int prow = r + half * 8;
        Pw[prow * 40 + ln]      = to_bf16(p0);
        Pw[prow * 40 + 16 + ln] = to_bf16(p1);
    }
    // ---- O += P * V^T ----
    v8bf p0v = *(const v8bf*)&Pw[ln * 40 + half * 8];
    v8bf p1v = *(const v8bf*)&Pw[ln * 40 + 16 + half * 8];
    v16bf pf = mk16(p0v, p1v);
#pragma unroll
    for (int t = 0; t < 8; ++t) {
        const int d = t * 16 + ln;
        const bf16_t* vp = vth + (size_t)d * S_LEN + j0 + half * 8;
        v8bf b0 = *(const v8bf*)vp;
        v8bf b1 = *(const v8bf*)(vp + 16);
        o[t] = wmma_bf16(pf, mk16(b0, b1), o[t]);
    }
}

__global__ __launch_bounds__(128)
void attn_kernel(const bf16_t* __restrict__ qb, const bf16_t* __restrict__ kb,
                 const bf16_t* __restrict__ vt, bf16_t* __restrict__ ctx) {
    __shared__ __align__(16) bf16_t P[4][16 * 40];

    const int hq   = blockIdx.x >> 5;
    const int qblk = blockIdx.x & 31;
    const int w = threadIdx.x >> 5, lane = threadIdx.x & 31;
    const int half = lane >> 4, ln = lane & 15;
    const int kvh = hq / GQA;
    const int i0 = qblk * 64 + w * 16;

    const bf16_t* kbh = kb + (size_t)kvh * S_LEN * HD;
    const bf16_t* vth = vt + (size_t)kvh * HD * S_LEN;
    bf16_t* Pw = &P[w][0];

    v16bf qf[4];
    {
        const bf16_t* qrow = qb + ((size_t)hq * S_LEN + (i0 + ln)) * HD;
#pragma unroll
        for (int kc = 0; kc < 4; ++kc) {
            v8bf a0 = *(const v8bf*)(qrow + kc * 32 + half * 8);
            v8bf a1 = *(const v8bf*)(qrow + kc * 32 + 16 + half * 8);
            qf[kc] = mk16(a0, a1);
        }
    }

    v8f o[8];
#pragma unroll
    for (int t = 0; t < 8; ++t)
#pragma unroll
        for (int r = 0; r < 8; ++r) o[t][r] = 0.0f;
    float mstat[8], lstat[8];
#pragma unroll
    for (int r = 0; r < 8; ++r) { mstat[r] = -1e30f; lstat[r] = 0.0f; }

    // full (unmasked) tiles, then exactly one diagonal (masked) tile
    const int nfull = (i0 + 1) >> 5;
    int j0 = 0;
    for (; j0 < nfull * 32; j0 += 32)
        attn_tile<false>(j0, i0, half, ln, kbh, vth, Pw, qf, o, mstat, lstat);
    attn_tile<true>(j0, i0, half, ln, kbh, vth, Pw, qf, o, mstat, lstat);

    // ---- normalize + store ctx bf16 [S][NH*HD] ----
#pragma unroll
    for (int t = 0; t < 8; ++t) {
        const int d = t * 16 + ln;
#pragma unroll
        for (int r = 0; r < 8; ++r) {
            const int i = i0 + r + half * 8;
            float val = o[t][r] / lstat[r];
            ctx[(size_t)i * (NH * HD) + hq * HD + d] = to_bf16(val);
        }
    }
}

// =====================================================================
// host launcher
// =====================================================================
extern "C" void kernel_launch(void* const* d_in, const int* in_sizes, int n_in,
                              void* d_out, int out_size, void* d_ws, size_t ws_size,
                              hipStream_t stream) {
    const int*   positions = (const int*)d_in[0];
    const float* hidden    = (const float*)d_in[1];   // [2048, 5120]
    const float* Wqkv      = (const float*)d_in[2];   // [7168, 5120]
    const float* Wo        = (const float*)d_in[3];   // [5120, 5120]
    const float* w_qk      = (const float*)d_in[4];   // [128]

    char* ws = (char*)d_ws;
    size_t off = 0;
    float*  qkv = (float*)(ws + off);  off += (size_t)S_LEN * QKVW * 4;
    bf16_t* qb  = (bf16_t*)(ws + off); off += (size_t)NH  * S_LEN * HD * 2;
    bf16_t* kb  = (bf16_t*)(ws + off); off += (size_t)NKV * S_LEN * HD * 2;
    bf16_t* vt  = (bf16_t*)(ws + off); off += (size_t)NKV * HD * S_LEN * 2;
    bf16_t* ctx = (bf16_t*)(ws + off); off += (size_t)S_LEN * NH * HD * 2;

    gemm_kernel<true><<<dim3(QKVW / 128, S_LEN / 128), 256, 0, stream>>>(
        (const void*)hidden, Wqkv, qkv, HDIM, QKVW);

    rope_norm_kernel<<<S_LEN, 256, 0, stream>>>(positions, qkv, w_qk, qb, kb, vt);

    attn_kernel<<<NH * (S_LEN / 64), 128, 0, stream>>>(qb, kb, vt, ctx);

    gemm_kernel<false><<<dim3(HDIM / 128, S_LEN / 128), 256, 0, stream>>>(
        (const void*)ctx, Wo, (float*)d_out, NH * HD, HDIM);
}
